// NodeFeatLayer_75952201663107
// MI455X (gfx1250) — compile-verified, matrix operand order checked by reading, not run on hardware
//
#include <hip/hip_runtime.h>

// ---------------- types ----------------
typedef __attribute__((ext_vector_type(16))) __bf16          bf16x16;
typedef __attribute__((ext_vector_type(8)))  float           f32x8;
typedef __attribute__((ext_vector_type(16))) unsigned short  u16x16;
typedef __attribute__((ext_vector_type(8)))  unsigned short  u16x8;

#define NDIM 256        // node_dim (K)
#define ODIM 256        // out_dim  (N of GEMM)
#define ROWS_PER_WG 64  // 4 M-tiles of 16
#define A_STRIDE 264    // ushorts per LDS row: 256 + 8 pad (conflict-free b128 reads)
#define KSTEPS (NDIM / 32)

__device__ __forceinline__ unsigned short f32_to_bf16(float f) {
  unsigned int u = __float_as_uint(f);
  unsigned int r = u + 0x7FFFu + ((u >> 16) & 1u);   // round-to-nearest-even
  return (unsigned short)(r >> 16);
}

// one-DS-op XOR butterfly step (group-of-32 swizzle: and=0x1F, or=0, xor=MASK)
template <int MASK>
__device__ __forceinline__ float swz_xor(float v) {
  return __int_as_float(
      __builtin_amdgcn_ds_swizzle(__float_as_int(v), (MASK << 10) | 0x1F));
}

// ---------------- kernel 0: pre-swizzle Wl into WMMA B-fragment order -------
// out[((ntile*8 + ks)*32 + lane)*16 + e] = bf16(Wl[k*256 + n])
//   k = ks*32 + (e&7) + ((e>>3)<<4) + ((lane>>4)<<3),  n = ntile*16 + (lane&15)
__global__ __launch_bounds__(256) void wl_swizzle_kernel(
    const float* __restrict__ Wl, unsigned short* __restrict__ out)
{
  int idx   = blockIdx.x * blockDim.x + threadIdx.x;   // 65536 total
  int e     = idx & 15;
  int lane  = (idx >> 4) & 31;
  int ks    = (idx >> 9) & 7;
  int ntile = idx >> 12;
  int k = ks * 32 + (e & 7) + ((e >> 3) << 4) + ((lane >> 4) << 3);
  int n = ntile * 16 + (lane & 15);
  out[idx] = f32_to_bf16(Wl[k * ODIM + n]);
}

// ---------------- kernel 1: FiLM params  gb = cond @ Wc + bc (gamma+=1) ------
__global__ __launch_bounds__(256) void film_params_kernel(
    const float* __restrict__ cond,   // [B,512]
    const float* __restrict__ Wc,     // [512,512]
    const float* __restrict__ bc,     // [512]
    float* __restrict__ gb,           // [B,512] (0..255 gamma+1, 256..511 beta)
    int total)
{
  int g = blockIdx.x * blockDim.x + threadIdx.x;
  if (g >= total) return;
  int b = g >> 9, c = g & 511;
  const float* cr = cond + b * 512;
  float s = 0.f;
  #pragma unroll 4
  for (int k = 0; k < 512; ++k) s = fmaf(cr[k], Wc[k * 512 + c], s);
  s += bc[c];
  if (c < 256) s += 1.0f;
  gb[g] = s;
}

// ---------------- kernel 2: x = relu(FiLM(LN(node_feats @ Wl))) --------------
// 512 threads = 16 waves. WG owns 64 rows x 256 cols. Wave w owns N-tile w and
// 4 M-tiles => each B fragment (2x global b128) reused by 4 WMMAs.
__global__ __launch_bounds__(512) void film_gemm_kernel(
    const float*          __restrict__ node_feats,  // [N,256]
    const unsigned short* __restrict__ Wlswz,       // pre-swizzled bf16 B frags
    const int*            __restrict__ batch_ids,   // [N]
    const float*          __restrict__ gb,          // [B,512]
    float*                __restrict__ x_out,       // [N,256]
    int N)
{
  __shared__ __align__(16) unsigned short A_lds[ROWS_PER_WG * A_STRIDE];
  __shared__ float row_sum[ROWS_PER_WG];
  __shared__ float row_sq [ROWS_PER_WG];
  __shared__ int   row_bid[ROWS_PER_WG];

  const int tid  = threadIdx.x;
  const int row0 = blockIdx.x * ROWS_PER_WG;

  if (tid < ROWS_PER_WG) {
    row_sum[tid] = 0.f;
    row_sq [tid] = 0.f;
    int gr = row0 + tid;
    row_bid[tid] = (gr < N) ? batch_ids[gr] : 0;
  }

  // stage A rows into LDS as bf16 (coalesced f32 loads, ds_store_b16)
  for (int i = tid; i < ROWS_PER_WG * NDIM; i += 512) {
    int r = i >> 8;
    int k = i & 255;
    int gr = row0 + r;
    float v = (gr < N) ? node_feats[gr * NDIM + k] : 0.f;
    A_lds[r * A_STRIDE + k] = f32_to_bf16(v);
  }
  __syncthreads();

  const int wave = tid >> 5;
  const int lane = tid & 31;
  const int hlf  = lane >> 4;          // 0 | 1
  const int l16  = lane & 15;
  const int ncol = wave * 16 + l16;    // column owned by this lane (B/C frags)

  f32x8 acc[4] = {};

  // per-wave base into pre-swizzled B: fragment for (ntile=wave, ks, lane)
  const unsigned short* wb = Wlswz + ((wave * KSTEPS) * 32 + lane) * 16;

  #pragma unroll
  for (int ks = 0; ks < KSTEPS; ++ks) {
    const int kb = ks * 32;
    // B fragment: 32 contiguous bytes per lane (2x global_load_b128)
    bf16x16 bf = __builtin_bit_cast(
        bf16x16, *(const u16x16*)(wb + ks * 32 * 16));

    #pragma unroll
    for (int mt = 0; mt < 4; ++mt) {
      // A fragment: lane holds row M = l16 of tile mt; two 16B LDS loads.
      const unsigned short* ap = &A_lds[(mt * 16 + l16) * A_STRIDE + kb + hlf * 8];
      u16x8 alo = *(const u16x8*)(ap);
      u16x8 ahi = *(const u16x8*)(ap + 16);
      u16x16 au;
      #pragma unroll
      for (int e = 0; e < 8; ++e) { au[e] = alo[e]; au[e + 8] = ahi[e]; }
      bf16x16 af = __builtin_bit_cast(bf16x16, au);

      acc[mt] = __builtin_amdgcn_wmma_f32_16x16x32_bf16(
          false, af, false, bf, (short)0, acc[mt], false, false);
    }
  }

  // LayerNorm partials: XOR butterfly over the 16 lanes of each half via
  // ds_swizzle (no clamp VALU), then LDS float atomics across the 16 waves.
  #pragma unroll
  for (int mt = 0; mt < 4; ++mt) {
    #pragma unroll
    for (int r = 0; r < 8; ++r) {
      float v = acc[mt][r];
      float s = v, q = v * v;
      s += swz_xor<1>(s);  q += swz_xor<1>(q);
      s += swz_xor<2>(s);  q += swz_xor<2>(q);
      s += swz_xor<4>(s);  q += swz_xor<4>(q);
      s += swz_xor<8>(s);  q += swz_xor<8>(q);
      if (l16 == 0) {
        int row = mt * 16 + r + hlf * 8;
        atomicAdd(&row_sum[row], s);
        atomicAdd(&row_sq [row], q);
      }
    }
  }
  __syncthreads();

  // normalize + FiLM + relu + store
  #pragma unroll
  for (int mt = 0; mt < 4; ++mt) {
    #pragma unroll
    for (int r = 0; r < 8; ++r) {
      int row  = mt * 16 + r + hlf * 8;
      int grow = row0 + row;
      if (grow < N) {
        float mu  = row_sum[row] * (1.f / 256.f);
        float var = row_sq [row] * (1.f / 256.f) - mu * mu;
        float rs  = rsqrtf(var + 1e-5f);
        int   b   = row_bid[row];
        float g   = gb[b * 512 + ncol];         // gamma + 1
        float be  = gb[b * 512 + 256 + ncol];   // beta
        float y   = (acc[mt][r] - mu) * rs * g + be;
        x_out[grow * ODIM + ncol] = fmaxf(y, 0.f);
      }
    }
  }
}

// ---------------- kernel 3: zero the accumulator ----------------------------
__global__ __launch_bounds__(256) void zero_kernel(float4* __restrict__ p, int n4) {
  int i = blockIdx.x * blockDim.x + threadIdx.x;
  if (i < n4) p[i] = float4{0.f, 0.f, 0.f, 0.f};
}

// ---------------- kernel 4: gather -> weight -> scatter-add (L2 atomics) ----
__global__ __launch_bounds__(256) void edge_kernel(
    const float* __restrict__ x,       // [N,256]
    const int*   __restrict__ jid,
    const int*   __restrict__ iid,
    const float* __restrict__ ew,
    const float* __restrict__ ep,
    float* __restrict__ out,           // [N,256] accumulator
    int E)
{
  int e = blockIdx.x * 8 + (threadIdx.x >> 5);   // one wave per edge
  if (e >= E) return;
  int lane = threadIdx.x & 31;
  int j = jid[e], i = iid[e];
  float w = ew[e] * ep[e];
  const float4* xs = (const float4*)(x + j * 256);
  float* ob = out + i * 256;
  #pragma unroll
  for (int h = 0; h < 2; ++h) {
    int c4 = lane + h * 32;
    float4 v = xs[c4];                           // coalesced 128b gather (L2-hot)
    int c = c4 * 4;
    atomicAdd(ob + c + 0, v.x * w);              // global_atomic_add_f32 (no rtn)
    atomicAdd(ob + c + 1, v.y * w);
    atomicAdd(ob + c + 2, v.z * w);
    atomicAdd(ob + c + 3, v.w * w);
  }
}

// ---------------- kernel 5: final relu (in place) ---------------------------
__global__ __launch_bounds__(256) void relu_kernel(float4* __restrict__ p, int n4) {
  int i = blockIdx.x * blockDim.x + threadIdx.x;
  if (i < n4) {
    float4 v = p[i];
    v.x = fmaxf(v.x, 0.f); v.y = fmaxf(v.y, 0.f);
    v.z = fmaxf(v.z, 0.f); v.w = fmaxf(v.w, 0.f);
    p[i] = v;
  }
}

// ---------------- host ------------------------------------------------------
extern "C" void kernel_launch(void* const* d_in, const int* in_sizes, int n_in,
                              void* d_out, int out_size, void* d_ws, size_t ws_size,
                              hipStream_t stream) {
  const float* node_feats = (const float*)d_in[0];
  const float* cond_feats = (const float*)d_in[1];
  const int*   batch_ids  = (const int*)  d_in[2];
  const int*   node_j     = (const int*)  d_in[3];
  const int*   node_i     = (const int*)  d_in[4];
  const float* edge_w     = (const float*)d_in[5];
  const float* edge_p     = (const float*)d_in[6];
  const float* Wc         = (const float*)d_in[7];
  const float* bc         = (const float*)d_in[8];
  const float* Wl         = (const float*)d_in[9];
  float* out = (float*)d_out;

  const int N = in_sizes[2];                 // 20000
  const int E = in_sizes[3];                 // 320000
  const int gb_total = in_sizes[1];          // B * 512 = 8192

  // workspace layout:
  //   [0, 32KB)        gb (FiLM params)
  //   [32KB, 160KB)    Wl pre-swizzled bf16 B fragments (65536 ushorts)
  //   [160KB, ...)     x buffer [N,256] f32
  float*          gbw   = (float*)d_ws;
  unsigned short* wlswz = (unsigned short*)((char*)d_ws + 32768);
  float*          xws   = (float*)((char*)d_ws + 32768 + 131072);

  wl_swizzle_kernel<<<65536 / 256, 256, 0, stream>>>(Wl, wlswz);

  film_params_kernel<<<(gb_total + 255) / 256, 256, 0, stream>>>(
      cond_feats, Wc, bc, gbw, gb_total);

  film_gemm_kernel<<<(N + ROWS_PER_WG - 1) / ROWS_PER_WG, 512, 0, stream>>>(
      node_feats, wlswz, batch_ids, gbw, xws, N);

  int n4 = (N * ODIM) / 4;
  zero_kernel<<<(n4 + 255) / 256, 256, 0, stream>>>((float4*)out, n4);

  edge_kernel<<<(E + 7) / 8, 256, 0, stream>>>(
      xws, node_j, node_i, edge_w, edge_p, out, E);

  relu_kernel<<<(n4 + 255) / 256, 256, 0, stream>>>((float4*)out, n4);
}
